// FeedForwardNetwork_6468220748602
// MI455X (gfx1250) — matrix operand order repaired and checked by reading, not compiled, not run on hardware
//
#include <hip/hip_runtime.h>
#include <hip/hip_bf16.h>

typedef __attribute__((ext_vector_type(16))) __bf16 v16bf;
typedef __attribute__((ext_vector_type(8)))  float  v8f;

#define D_IN   1920
#define TILE_M 32            // nodes per workgroup (2 row-tiles of 16)
#define NWAVE  16            // 512 threads

// padded LDS leading dimensions: ld*2 bytes == 16 (mod 256) -> conflict-free b128 A-loads
#define LDX0 520             // x0 rows (512 + 8)
#define LDX1 264             // x1 component rows (256 + 8)
#define LDX2 136             // x2 component rows (128 + 8)
#define LDG  1160            // gate rows (1152 + 8)
#define LDS_ 1544            // silu(s) rows (1536 + 8)
#define LDV1 776             // gated v1 rows (768 + 8)
#define LDV2 392             // gated v2 rows (384 + 8)

#define XREG (3 * TILE_M * LDX1)   // 25344 elems: fits x0 (32*520), 3x x1, 5x x2 (32*136)
#define UREG (3 * TILE_M * LDV1)   // 74496 elems: fits s (32*1544), 3x v1g, 5x v2g

// ---------------- WMMA fragment loaders (wave32, 16x16x32 bf16) ----------------
static __device__ __forceinline__ v16bf frag_a_lds(const __bf16* X, int ldx, int kb, int lane) {
    const int row = lane & 15;
    const int kh  = (lane >> 4) & 1;
    const __bf16* p = X + row * ldx + kb + kh * 8;
    union { uint4 q[2]; v16bf v; } u;
    u.q[0] = *reinterpret_cast<const uint4*>(p);
    u.q[1] = *reinterpret_cast<const uint4*>(p + 16);
    return u.v;
}

// B from fragment-packed weights: fragment (ct, kb32) = 1KB contiguous block,
// lane's 16 bf16 at block + lane*32B -> one fully coalesced wave read.
static __device__ __forceinline__ v16bf frag_b_pk(const __bf16* Wp, int nkb, int ct, int kb32, int lane) {
    const __bf16* p = Wp + ((long)(ct * nkb + kb32) << 9) + lane * 16;
    union { uint4 q[2]; v16bf v; } u;
    u.q[0] = *reinterpret_cast<const uint4*>(p);
    u.q[1] = *reinterpret_cast<const uint4*>(p + 8);
    return u.v;
}

static __device__ __forceinline__ v8f wmma_bf16(v16bf a, v16bf b, v8f c) {
    return __builtin_amdgcn_wmma_f32_16x16x32_bf16(false, a, false, b, (short)0, c, false, false);
}

// ---------------- weight f32 -> bf16 fragment packing ----------------
__global__ void pack_w(const float* __restrict__ src, __bf16* __restrict__ dst, int O, int C) {
    int idx = blockIdx.x * 256 + threadIdx.x;
    int nkb = C >> 5;
    int total = (O >> 4) * nkb * 32;
    if (idx >= total) return;
    int lane = idx & 31;
    int f    = idx >> 5;
    int kb32 = f % nkb;
    int ct   = f / nkb;
    const float* s = src + (long)(ct * 16 + (lane & 15)) * C + kb32 * 32 + (lane >> 4) * 8;
    __bf16* d = dst + ((long)f << 9) + lane * 16;
    #pragma unroll
    for (int e = 0; e < 16; ++e) d[e] = (__bf16)s[(e & 7) + ((e >> 3) << 4)];
}

// ---------------- fused two-stage equivariant FFN ----------------
__global__ __launch_bounds__(512)
void ffn_fused(const float* __restrict__ x, const float* __restrict__ attr,
               const float* __restrict__ b1s, const float* __restrict__ b1g,
               const float* __restrict__ b2s,
               const __bf16* __restrict__ w1s,  const __bf16* __restrict__ w1g,
               const __bf16* __restrict__ w1v1, const __bf16* __restrict__ w1v2,
               const __bf16* __restrict__ w2s,  const __bf16* __restrict__ w2v1,
               const __bf16* __restrict__ w2v2,
               float* __restrict__ out)
{
    __shared__ __bf16 sX[XREG];
    __shared__ __bf16 sG[TILE_M * LDG];
    __shared__ __bf16 sU[UREG];
    __shared__ float  sA[TILE_M];

    const int tid   = threadIdx.x;
    const int wid   = tid >> 5;
    const int lane  = tid & 31;
    const int nn    = lane & 15;
    const int rbase = (lane >> 4) << 3;
    const int nodeBase = blockIdx.x * TILE_M;

    if (tid < TILE_M) sA[tid] = attr[nodeBase + tid];
    for (int i = tid; i < TILE_M * 512; i += 512) {
        int r = i >> 9, k = i & 511;
        sX[r * LDX0 + k] = (__bf16)x[(long)(nodeBase + r) * D_IN + k];
    }
    __syncthreads();

    // ===== stage 1 scalar: s = silu(.), g = sigmoid(.)   168 jobs, 2 row-tiles share B =====
    for (int t = wid; t < 168; t += NWAVE) {
        const bool isS = (t < 96);
        const int  ct  = isS ? t : t - 96;
        const int  colbase = ct * 16;
        const __bf16* Wp  = isS ? w1s : w1g;
        const float*  bia = isS ? b1s : b1g;
        v8f acc[2] = {};
        #pragma unroll 4
        for (int kb32 = 0; kb32 < 16; ++kb32) {
            v16bf b = frag_b_pk(Wp, 16, ct, kb32, lane);
            #pragma unroll
            for (int rt = 0; rt < 2; ++rt)
                acc[rt] = wmma_bf16(frag_a_lds(sX + rt * 16 * LDX0, LDX0, kb32 * 32, lane), b, acc[rt]);
        }
        const float bv = bia[colbase + nn];
        const float sc = 0.04419417382415922f;   // 1/sqrt(512)
        #pragma unroll
        for (int rt = 0; rt < 2; ++rt)
        #pragma unroll
        for (int j = 0; j < 8; ++j) {
            const int row = rt * 16 + rbase + j;
            float v  = acc[rt][j] * sc * sA[row] + bv;
            float sg = 1.f / (1.f + __expf(-v));
            if (isS) sU[row * LDS_ + colbase + nn] = (__bf16)(v * sg);
            else     sG[row * LDG  + colbase + nn] = (__bf16)sg;
        }
    }
    __syncthreads();

    // ===== o0 = silu(s) @ W2_s^T   32 jobs, K = 1536 =====
    for (int t = wid; t < 32; t += NWAVE) {
        const int colbase = t * 16;
        v8f acc[2] = {};
        #pragma unroll 4
        for (int kb32 = 0; kb32 < 48; ++kb32) {
            v16bf b = frag_b_pk(w2s, 48, t, kb32, lane);
            #pragma unroll
            for (int rt = 0; rt < 2; ++rt)
                acc[rt] = wmma_bf16(frag_a_lds(sU + rt * 16 * LDS_, LDS_, kb32 * 32, lane), b, acc[rt]);
        }
        const float bv = b2s[colbase + nn];
        const float sc = 0.02551551815399144f;   // 1/sqrt(1536)
        #pragma unroll
        for (int rt = 0; rt < 2; ++rt)
        #pragma unroll
        for (int j = 0; j < 8; ++j) {
            const int row = rt * 16 + rbase + j;
            out[(long)(nodeBase + row) * D_IN + colbase + nn] = acc[rt][j] * sc * sA[row] + bv;
        }
    }

    // stage x1 (all 3 components; contiguous global reads) -- x0 dead after stage 1
    for (int i = tid; i < TILE_M * 256 * 3; i += 512) {
        int m = i % 3; int c = (i / 3) & 255; int r = i / 768;
        sX[m * (TILE_M * LDX1) + r * LDX1 + c] = (__bf16)x[(long)(nodeBase + r) * D_IN + 512 + c * 3 + m];
    }
    __syncthreads();   // o0 done reading sU; x1 staged

    // ===== v1g: (x1m @ W1_v1^T)*sc*a*g    48 jobs, 6 accumulators share B =====
    for (int t = wid; t < 48; t += NWAVE) {
        const int colbase = t * 16;
        v8f acc[6] = {};
        #pragma unroll 2
        for (int kb32 = 0; kb32 < 8; ++kb32) {
            v16bf b = frag_b_pk(w1v1, 8, t, kb32, lane);
            #pragma unroll
            for (int rt = 0; rt < 2; ++rt)
            #pragma unroll
            for (int m = 0; m < 3; ++m)
                acc[rt * 3 + m] = wmma_bf16(
                    frag_a_lds(sX + m * (TILE_M * LDX1) + rt * 16 * LDX1, LDX1, kb32 * 32, lane),
                    b, acc[rt * 3 + m]);
        }
        const float sc = 0.0625f;                // 1/sqrt(256)
        #pragma unroll
        for (int rt = 0; rt < 2; ++rt)
        #pragma unroll
        for (int m = 0; m < 3; ++m)
        #pragma unroll
        for (int j = 0; j < 8; ++j) {
            const int row = rt * 16 + rbase + j;
            float g = (float)sG[row * LDG + colbase + nn];
            sU[m * (TILE_M * LDV1) + row * LDV1 + colbase + nn] =
                (__bf16)(acc[rt * 3 + m][j] * sc * sA[row] * g);
        }
    }
    __syncthreads();

    // stage x2 (all 5 components) -- x1 dead after v1g
    for (int i = tid; i < TILE_M * 128 * 5; i += 512) {
        int m = i % 5; int c = (i / 5) & 127; int r = i / 640;
        sX[m * (TILE_M * LDX2) + r * LDX2 + c] = (__bf16)x[(long)(nodeBase + r) * D_IN + 1280 + c * 5 + m];
    }

    // ===== o1: v1g @ W2_v1^T    16 jobs, K = 768 =====
    for (int t = wid; t < 16; t += NWAVE) {
        const int colbase = t * 16;
        v8f acc[6] = {};
        #pragma unroll 2
        for (int kb32 = 0; kb32 < 24; ++kb32) {
            v16bf b = frag_b_pk(w2v1, 24, t, kb32, lane);
            #pragma unroll
            for (int rt = 0; rt < 2; ++rt)
            #pragma unroll
            for (int m = 0; m < 3; ++m)
                acc[rt * 3 + m] = wmma_bf16(
                    frag_a_lds(sU + m * (TILE_M * LDV1) + rt * 16 * LDV1, LDV1, kb32 * 32, lane),
                    b, acc[rt * 3 + m]);
        }
        const float sc = 0.03608439182435161f;   // 1/sqrt(768)
        #pragma unroll
        for (int rt = 0; rt < 2; ++rt)
        #pragma unroll
        for (int m = 0; m < 3; ++m)
        #pragma unroll
        for (int j = 0; j < 8; ++j) {
            const int row = rt * 16 + rbase + j;
            out[(long)(nodeBase + row) * D_IN + 512 + (colbase + nn) * 3 + m] =
                acc[rt * 3 + m][j] * sc * sA[row];
        }
    }
    __syncthreads();   // o1 done reading sU; x2 staged

    // ===== v2g: (x2m @ W1_v2^T)*sc*a*g    48 jobs (ct x rt), 5 accumulators share B =====
    for (int t = wid; t < 48; t += NWAVE) {
        const int rt = t & 1;
        const int ct = t >> 1;
        const int colbase = ct * 16;
        v8f acc[5] = {};
        #pragma unroll 2
        for (int kb32 = 0; kb32 < 4; ++kb32) {
            v16bf b = frag_b_pk(w1v2, 4, ct, kb32, lane);
            #pragma unroll
            for (int m = 0; m < 5; ++m)
                acc[m] = wmma_bf16(
                    frag_a_lds(sX + m * (TILE_M * LDX2) + rt * 16 * LDX2, LDX2, kb32 * 32, lane),
                    b, acc[m]);
        }
        const float sc = 0.08838834764831845f;   // 1/sqrt(128)
        #pragma unroll
        for (int m = 0; m < 5; ++m)
        #pragma unroll
        for (int j = 0; j < 8; ++j) {
            const int row = rt * 16 + rbase + j;
            float g = (float)sG[row * LDG + 768 + colbase + nn];
            sU[m * (TILE_M * LDV2) + row * LDV2 + colbase + nn] =
                (__bf16)(acc[m][j] * sc * sA[row] * g);
        }
    }
    __syncthreads();

    // ===== o2: v2g @ W2_v2^T    16 jobs (ct x rt), K = 384 =====
    for (int t = wid; t < 16; t += NWAVE) {
        const int rt = t & 1;
        const int ct = t >> 1;
        const int colbase = ct * 16;
        v8f acc[5] = {};
        #pragma unroll 2
        for (int kb32 = 0; kb32 < 12; ++kb32) {
            v16bf b = frag_b_pk(w2v2, 12, ct, kb32, lane);
            #pragma unroll
            for (int m = 0; m < 5; ++m)
                acc[m] = wmma_bf16(
                    frag_a_lds(sU + m * (TILE_M * LDV2) + rt * 16 * LDV2, LDV2, kb32 * 32, lane),
                    b, acc[m]);
        }
        const float sc = 0.05103103630798288f;   // 1/sqrt(384)
        #pragma unroll
        for (int m = 0; m < 5; ++m)
        #pragma unroll
        for (int j = 0; j < 8; ++j) {
            const int row = rt * 16 + rbase + j;
            out[(long)(nodeBase + row) * D_IN + 1280 + (colbase + nn) * 5 + m] =
                acc[m][j] * sc * sA[row];
        }
    }
}

extern "C" void kernel_launch(void* const* d_in, const int* in_sizes, int n_in,
                              void* d_out, int out_size, void* d_ws, size_t ws_size,
                              hipStream_t stream) {
    const float* x    = (const float*)d_in[0];
    const float* attr = (const float*)d_in[1];
    const float* W1s  = (const float*)d_in[2];
    const float* W1g  = (const float*)d_in[3];
    const float* W1v1 = (const float*)d_in[4];
    const float* W1v2 = (const float*)d_in[5];
    const float* b1s  = (const float*)d_in[6];
    const float* b1g  = (const float*)d_in[7];
    const float* W2s  = (const float*)d_in[8];
    const float* W2v1 = (const float*)d_in[9];
    const float* W2v2 = (const float*)d_in[10];
    const float* b2s  = (const float*)d_in[11];
    float* out = (float*)d_out;

    __bf16* ws   = (__bf16*)d_ws;
    __bf16* w1s  = ws;                      // 1536*512
    __bf16* w1g  = w1s  + 1536 * 512;       // 1152*512
    __bf16* w1v1 = w1g  + 1152 * 512;       //  768*256
    __bf16* w1v2 = w1v1 +  768 * 256;       //  384*128
    __bf16* w2s  = w1v2 +  384 * 128;       //  512*1536
    __bf16* w2v1 = w2s  +  512 * 1536;      //  256*768
    __bf16* w2v2 = w2v1 +  256 * 768;       //  128*384

    struct { const float* s; __bf16* d; int O, C; } pk[7] = {
        {W1s,  w1s,  1536, 512}, {W1g,  w1g,  1152, 512},
        {W1v1, w1v1,  768, 256}, {W1v2, w1v2,  384, 128},
        {W2s,  w2s,   512, 1536},{W2v1, w2v1,  256, 768},
        {W2v2, w2v2,  128, 384},
    };
    for (int i = 0; i < 7; ++i) {
        int total = (pk[i].O >> 4) * (pk[i].C >> 5) * 32;
        pack_w<<<(total + 255) / 256, 256, 0, stream>>>(pk[i].s, pk[i].d, pk[i].O, pk[i].C);
    }

    ffn_fused<<<16384 / TILE_M, 512, 0, stream>>>(x, attr, b1s, b1g, b2s,
                                                  w1s, w1g, w1v1, w1v2,
                                                  w2s, w2v1, w2v2, out);
}
